// CrossLayerKVAttention_65687229825402
// MI455X (gfx1250) — matrix operand order repaired and checked by reading, not compiled, not run on hardware
//
#include <hip/hip_runtime.h>
#include <hip/hip_bf16.h>

typedef __bf16 bf16_t;
typedef __attribute__((ext_vector_type(16))) __bf16 v16bf;
typedef __attribute__((ext_vector_type(8)))  __bf16 v8bf;
typedef __attribute__((ext_vector_type(4)))  __bf16 v4bf;
typedef __attribute__((ext_vector_type(8)))  float  v8f;
typedef __attribute__((ext_vector_type(4)))  float  v4f;

#define NH_C  16
#define NKV_C 8
#define HD_C  128

__device__ __forceinline__ v8f wmma_bf16(v16bf a, v16bf b, v8f c) {
    return __builtin_amdgcn_wmma_f32_16x16x32_bf16(false, a, false, b, (short)0, c,
                                                   false, false);
}

// Max-reduce over each 16-lane half (xor butterfly). VALU permlane16 if
// available (no LDS pipe, no dscnt waits), else ds_bpermute via __shfl_xor.
__device__ __forceinline__ float half16_reduce_max(float x) {
#if __has_builtin(__builtin_amdgcn_permlane16)
#define PL16(s0, s1)                                                          \
    __builtin_bit_cast(float, __builtin_amdgcn_permlane16(                    \
        __builtin_bit_cast(unsigned, x), __builtin_bit_cast(unsigned, x),     \
        (s0), (s1), false, false))
    x = fmaxf(x, PL16(0x67452301u, 0xEFCDAB89u));   // xor 1
    x = fmaxf(x, PL16(0x54761032u, 0xDCFE98BAu));   // xor 2
    x = fmaxf(x, PL16(0x32107654u, 0xBA98FEDCu));   // xor 4
    x = fmaxf(x, PL16(0xFEDCBA98u, 0x76543210u));   // xor 8
#undef PL16
#else
    x = fmaxf(x, __shfl_xor(x, 1, 32));
    x = fmaxf(x, __shfl_xor(x, 2, 32));
    x = fmaxf(x, __shfl_xor(x, 4, 32));
    x = fmaxf(x, __shfl_xor(x, 8, 32));
#endif
    return x;
}

// ---------------------------------------------------------------------------
// fp32 -> bf16 elementwise convert (n must be a multiple of 4)
// ---------------------------------------------------------------------------
__global__ void cvt_f32_bf16_kernel(const float* __restrict__ in,
                                    bf16_t* __restrict__ out, size_t n4)
{
    size_t i = (size_t)blockIdx.x * blockDim.x + threadIdx.x;
    if (i >= n4) return;
    v4f v = ((const v4f*)in)[i];
    v4bf o;
#pragma unroll
    for (int k = 0; k < 4; ++k) o[k] = (bf16_t)v[k];
    ((v4bf*)out)[i] = o;
}

// ---------------------------------------------------------------------------
// GEMM:  C[M,N] = A[M,K] * B[N,K]^T   (A, B bf16; C bf16 or fp32)
// Block = 128 threads (4 waves). Wave computes a 16(M) x 64(N) tile.
// All fragments preloaded into distinct registers before the WMMA batch.
// ---------------------------------------------------------------------------
template <bool OUT_BF16>
__global__ __launch_bounds__(128) void gemm_bb_wmma(
    const bf16_t* __restrict__ A, const bf16_t* __restrict__ Bw,
    void* __restrict__ Cptr, int M, int N, int K)
{
    const int wave  = threadIdx.x >> 5;
    const int lane  = threadIdx.x & 31;
    const int lhalf = lane >> 4;
    const int l15   = lane & 15;
    const int m0    = blockIdx.y * 64 + wave * 16;
    const int n0    = blockIdx.x * 64;

    const bf16_t* Ar = A + (size_t)(m0 + l15) * K;

    v8f acc[4] = {};

#pragma unroll 2
    for (int k0 = 0; k0 < K; k0 += 32) {
        // A fragment: 16x32, lane = row, K halves split by lane-half
        v16bf afrag;
        {
            const int abase = k0 + lhalf * 16;
            v8bf lo = *(const v8bf*)(Ar + abase);
            v8bf hi = *(const v8bf*)(Ar + abase + 8);
#pragma unroll
            for (int i = 0; i < 8; ++i) { afrag[i] = lo[i]; afrag[8 + i] = hi[i]; }
        }
        // 4 B fragments (32x16): lane = col, 16 consecutive K values
        v16bf bfrag[4];
#pragma unroll
        for (int nt = 0; nt < 4; ++nt)
            bfrag[nt] = *(const v16bf*)(Bw + (size_t)(n0 + nt * 16 + l15) * K
                                        + k0 + lhalf * 16);
#pragma unroll
        for (int nt = 0; nt < 4; ++nt)
            acc[nt] = wmma_bf16(afrag, bfrag[nt], acc[nt]);
    }

#pragma unroll
    for (int nt = 0; nt < 4; ++nt) {
        const int col = n0 + nt * 16 + l15;
#pragma unroll
        for (int j = 0; j < 8; ++j) {
            const int row = m0 + j + lhalf * 8;
            if (OUT_BF16)
                ((bf16_t*)Cptr)[(size_t)row * N + col] = (bf16_t)acc[nt][j];
            else
                ((float*)Cptr)[(size_t)row * N + col] = acc[nt][j];
        }
    }
}

// ---------------------------------------------------------------------------
// Transpose V: Vb [B*T, NKV*HD] (bf16) -> Vt [B, NKV, HD, T] (bf16)
// ---------------------------------------------------------------------------
__global__ void transpose_v_kernel(const bf16_t* __restrict__ Vb,
                                   bf16_t* __restrict__ Vt, int T)
{
    size_t idx = (size_t)blockIdx.x * blockDim.x + threadIdx.x;
    size_t total = (size_t)2 * T * (NKV_C * HD_C);
    if (idx >= total) return;
    int d      = (int)(idx & (HD_C - 1));
    size_t r   = idx >> 7;
    int kvh    = (int)(r & (NKV_C - 1));
    size_t bt  = r >> 3;
    int t      = (int)(bt % T);
    int b      = (int)(bt / T);
    Vt[(((size_t)(b * NKV_C + kvh) * HD_C + d) * T) + t] = Vb[idx];
}

// ---------------------------------------------------------------------------
// Pack kv_state (fp32): out [B, NKV, T, 2*HD] = concat(K, V)
// ---------------------------------------------------------------------------
__global__ void pack_kv_kernel(const bf16_t* __restrict__ Kb,
                               const bf16_t* __restrict__ Vb,
                               float* __restrict__ out, int T)
{
    size_t idx = (size_t)blockIdx.x * blockDim.x + threadIdx.x;
    size_t total = (size_t)2 * NKV_C * T * (2 * HD_C);
    if (idx >= total) return;
    int c     = (int)(idx & (2 * HD_C - 1));
    size_t r  = idx >> 8;
    int t     = (int)(r % T);
    size_t bk = r / T;
    int kvh   = (int)(bk & (NKV_C - 1));
    int b     = (int)(bk >> 3);
    size_t src = (size_t)(b * T + t) * (NKV_C * HD_C) + (size_t)kvh * HD_C;
    float v = (c < HD_C) ? (float)Kb[src + c] : (float)Vb[src + (c - HD_C)];
    out[idx] = v;
}

// ---------------------------------------------------------------------------
// Causal flash attention (GQA).  Block = 128 threads (4 waves).
// Wave handles 16 query rows of one (b, h); keys in blocks of 32.
// Row-max via permlane16 butterfly; row-sum via WMMA against all-ones.
// ---------------------------------------------------------------------------
__global__ __launch_bounds__(128) void attn_kernel(
    const bf16_t* __restrict__ Qb,   // [B*T, NH*HD]
    const bf16_t* __restrict__ Kb,   // [B*T, NKV*HD]
    const bf16_t* __restrict__ Vt,   // [B, NKV, HD, T]
    bf16_t* __restrict__ Ab,         // [B*T, NH*HD]
    int T)
{
    __shared__ bf16_t lds[4][16 * 32];

    const int wave  = threadIdx.x >> 5;
    const int lane  = threadIdx.x & 31;
    const int lhalf = lane >> 4;
    const int l15   = lane & 15;
    const int b     = blockIdx.z;
    const int h     = blockIdx.y;
    const int kv    = h >> 1;               // groups = NH/NKV = 2
    const int q0    = blockIdx.x * 64 + wave * 16;
    const float scale = 0.08838834764831845f;   // 1/sqrt(128)

    // all-ones B fragment for row-sum WMMA
    v16bf ones;
#pragma unroll
    for (int i = 0; i < 16; ++i) ones[i] = (bf16_t)1.0f;

    // Q fragments (A layout), hd = 4 chunks of 32
    v16bf qfrag[4];
    {
        const bf16_t* qr = Qb + (size_t)(b * T + q0 + l15) * (NH_C * HD_C) + h * HD_C;
#pragma unroll
        for (int s = 0; s < 4; ++s) {
            const int base = s * 32 + lhalf * 16;
            v8bf lo = *(const v8bf*)(qr + base);
            v8bf hi = *(const v8bf*)(qr + base + 8);
#pragma unroll
            for (int i = 0; i < 8; ++i) { qfrag[s][i] = lo[i]; qfrag[s][8 + i] = hi[i]; }
        }
    }

    float mrow[8], lrow[8];
    v8f o[8] = {};
#pragma unroll
    for (int j = 0; j < 8; ++j) { mrow[j] = -1e30f; lrow[j] = 0.0f; }

    bf16_t* myLds = lds[wave];
    const bf16_t* vbase = Vt + (size_t)(b * NKV_C + kv) * HD_C * T;
    const int kmax = q0 + 15;

    for (int kb = 0; kb * 32 <= kmax; ++kb) {
        const int kbase = kb * 32;

        // ---- preload K fragments, then batch the S WMMAs
        v16bf kf[2][4];
#pragma unroll
        for (int nt = 0; nt < 2; ++nt) {
            const bf16_t* kr = Kb + (size_t)(b * T + kbase + nt * 16 + l15)
                             * (NKV_C * HD_C) + kv * HD_C + lhalf * 16;
#pragma unroll
            for (int s = 0; s < 4; ++s) kf[nt][s] = *(const v16bf*)(kr + s * 32);
        }
        v8f sacc[2] = {};
#pragma unroll
        for (int nt = 0; nt < 2; ++nt)
#pragma unroll
            for (int s = 0; s < 4; ++s)
                sacc[nt] = wmma_bf16(qfrag[s], kf[nt][s], sacc[nt]);

        // ---- issue V fragment loads early (independent of softmax math)
        v16bf vf[8];
#pragma unroll
        for (int ht = 0; ht < 8; ++ht)
            vf[ht] = *(const v16bf*)(vbase + (size_t)(ht * 16 + l15) * T
                                     + kbase + lhalf * 16);

        // ---- scale + causal mask + online max (permlane butterfly per row)
        float p0v[8], p1v[8], corr[8];
#pragma unroll
        for (int j = 0; j < 8; ++j) {
            const int row = q0 + j + lhalf * 8;
            const int c0 = kbase + l15;
            const int c1 = kbase + 16 + l15;
            float v0 = (c0 <= row) ? sacc[0][j] * scale : -1e30f;
            float v1 = (c1 <= row) ? sacc[1][j] * scale : -1e30f;
            float mx = half16_reduce_max(fmaxf(v0, v1));
            const float mn = fmaxf(mrow[j], mx);
            p0v[j] = __expf(v0 - mn);
            p1v[j] = __expf(v1 - mn);
            corr[j] = __expf(mrow[j] - mn);
            mrow[j] = mn;
        }

        // ---- P (C layout) -> LDS -> P (A layout)
#pragma unroll
        for (int j = 0; j < 8; ++j) {
            const int r = j + lhalf * 8;
            myLds[r * 32 + l15]      = (bf16_t)p0v[j];
            myLds[r * 32 + 16 + l15] = (bf16_t)p1v[j];
        }
        asm volatile("s_wait_dscnt 0x0" ::: "memory");
        v16bf pfrag;
        {
            const bf16_t* pr = myLds + l15 * 32 + lhalf * 16;
#pragma unroll
            for (int i = 0; i < 8; ++i) { pfrag[i] = pr[i]; pfrag[8 + i] = pr[8 + i]; }
        }
        asm volatile("" ::: "memory");

        // ---- row sums with one WMMA against all-ones B
        v8f sums = {};
        sums = wmma_bf16(pfrag, ones, sums);

        // ---- update l, rescale running O, accumulate P @ V
#pragma unroll
        for (int j = 0; j < 8; ++j) lrow[j] = lrow[j] * corr[j] + sums[j];
#pragma unroll
        for (int ht = 0; ht < 8; ++ht)
#pragma unroll
            for (int j = 0; j < 8; ++j) o[ht][j] *= corr[j];
#pragma unroll
        for (int ht = 0; ht < 8; ++ht)
            o[ht] = wmma_bf16(pfrag, vf[ht], o[ht]);
    }

    // ---- normalize and store attn output (bf16)
#pragma unroll
    for (int ht = 0; ht < 8; ++ht) {
        const int col = h * HD_C + ht * 16 + l15;
#pragma unroll
        for (int j = 0; j < 8; ++j) {
            const int row = q0 + j + lhalf * 8;
            const float val = o[ht][j] / lrow[j];
            Ab[(size_t)(b * T + row) * (NH_C * HD_C) + col] = (bf16_t)val;
        }
    }
}

// ---------------------------------------------------------------------------
extern "C" void kernel_launch(void* const* d_in, const int* in_sizes, int n_in,
                              void* d_out, int out_size, void* d_ws, size_t ws_size,
                              hipStream_t stream) {
    const float* x  = (const float*)d_in[0];
    const float* Wq = (const float*)d_in[1];
    const float* Wk = (const float*)d_in[2];
    const float* Wv = (const float*)d_in[3];
    const float* Wo = (const float*)d_in[4];

    const int B = 2, T = 2048, D = 2048, KV = NKV_C * HD_C;  // KV = 1024
    const int M = B * T;                                      // 4096

    char* ws = (char*)d_ws;
    bf16_t* xb  = (bf16_t*)ws; ws += (size_t)M * D   * sizeof(bf16_t);
    bf16_t* Wqb = (bf16_t*)ws; ws += (size_t)D * D   * sizeof(bf16_t);
    bf16_t* Wkb = (bf16_t*)ws; ws += (size_t)KV * D  * sizeof(bf16_t);
    bf16_t* Wvb = (bf16_t*)ws; ws += (size_t)KV * D  * sizeof(bf16_t);
    bf16_t* Wob = (bf16_t*)ws; ws += (size_t)D * D   * sizeof(bf16_t);
    bf16_t* Qb  = (bf16_t*)ws; ws += (size_t)M * D   * sizeof(bf16_t);
    bf16_t* Kb  = (bf16_t*)ws; ws += (size_t)M * KV  * sizeof(bf16_t);
    bf16_t* Vb  = (bf16_t*)ws; ws += (size_t)M * KV  * sizeof(bf16_t);
    bf16_t* Vt  = (bf16_t*)ws; ws += (size_t)M * KV  * sizeof(bf16_t);
    bf16_t* Ab  = (bf16_t*)ws; ws += (size_t)M * D   * sizeof(bf16_t);

    float* out = (float*)d_out;                  // [B, T, D]
    float* kvs = out + (size_t)M * D;            // [B, NKV, T, 2*HD]

    dim3 blk(128);

    // one-time bf16 conversion of activations + weights
    auto cvt = [&](const float* src, bf16_t* dst, size_t n) {
        size_t n4 = n / 4;
        cvt_f32_bf16_kernel<<<(unsigned)((n4 + 255) / 256), 256, 0, stream>>>(src, dst, n4);
    };
    cvt(x,  xb,  (size_t)M * D);
    cvt(Wq, Wqb, (size_t)D * D);
    cvt(Wk, Wkb, (size_t)KV * D);
    cvt(Wv, Wvb, (size_t)KV * D);
    cvt(Wo, Wob, (size_t)D * D);

    // Q/K/V projections (bf16 in, bf16 out)
    gemm_bb_wmma<true><<<dim3(D / 64,  M / 64), blk, 0, stream>>>(xb, Wqb, Qb, M, D,  D);
    gemm_bb_wmma<true><<<dim3(KV / 64, M / 64), blk, 0, stream>>>(xb, Wkb, Kb, M, KV, D);
    gemm_bb_wmma<true><<<dim3(KV / 64, M / 64), blk, 0, stream>>>(xb, Wvb, Vb, M, KV, D);

    // V transpose for P@V operand layout
    {
        size_t total = (size_t)M * KV;
        transpose_v_kernel<<<(unsigned)((total + 255) / 256), 256, 0, stream>>>(Vb, Vt, T);
    }
    // kv_state output (fp32)
    {
        size_t total = (size_t)B * NKV_C * T * (2 * HD_C);
        pack_kv_kernel<<<(unsigned)((total + 255) / 256), 256, 0, stream>>>(Kb, Vb, kvs, T);
    }

    // causal GQA flash attention
    attn_kernel<<<dim3(T / 64, NH_C, B), blk, 0, stream>>>(Qb, Kb, Vt, Ab, T);

    // output projection (bf16 in, fp32 out) straight into d_out
    gemm_bb_wmma<false><<<dim3(D / 64, M / 64), blk, 0, stream>>>(Ab, Wob, out, M, D, D);
}
